// clipSCPFlow3D_28381143892158
// MI455X (gfx1250) — matrix operand order repaired and checked by reading, not compiled, not run on hardware
//
#include <hip/hip_runtime.h>
#include <hip/hip_bf16.h>

typedef __attribute__((ext_vector_type(16))) __bf16 v16bf;
typedef __attribute__((ext_vector_type(8)))  __bf16 v8bf;
typedef __attribute__((ext_vector_type(8)))  float  v8f;

#define DIM     80
#define CCH     32
#define N_CLS   36
#define NTILE_W 5            // 80 / 16
#define NW      4            // waves per block
#define NTILES  (DIM*DIM*NTILE_W)   // 32000

// ---------------------------------------------------------------------------
// Pack conv weights into WMMA B-operand layout (bf16), one v16bf per lane
// per tap.  B is K(=32 in-ch) x N(=16, out-ch 0..2 used).
// 16-bit B layout: lane L -> n = L%16, k = (L/16)*8 + s for s<8,
// k = 16 + (L/16)*8 + (s-8) for s>=8; slots s=0..15 packed two per VGPR.
// ---------------------------------------------------------------------------
__global__ void pack_b_kernel(const float* __restrict__ conv_w,
                              __bf16* __restrict__ Bops) {
    int i = blockIdx.x * blockDim.x + threadIdx.x;  // 27*32*16 = 13824
    if (i >= 27 * 32 * 16) return;
    int t    = i >> 9;          // tap 0..26  (kd*9 + kh*3 + kw)
    int lane = (i >> 4) & 31;
    int s    = i & 15;
    int n  = lane & 15;
    int hi = lane >> 4;
    int k  = (s < 8) ? hi * 8 + s : 16 + hi * 8 + (s - 8);       // in-channel
    float v = 0.f;
    if (n < 3) v = conv_w[(n * 32 + k) * 27 + t];
    Bops[i] = (__bf16)v;
}

// ---------------------------------------------------------------------------
// Tiny MLP: (36x768) -> relu 64 -> relu 128 -> 32.  One block, negligible cost.
// ---------------------------------------------------------------------------
__global__ void mlp_kernel(const float* __restrict__ tf,
                           const float* __restrict__ w1, const float* __restrict__ b1,
                           const float* __restrict__ w2, const float* __restrict__ b2,
                           const float* __restrict__ w3, const float* __restrict__ b3,
                           float* __restrict__ baseOut /* 36*32 */) {
    __shared__ float h1[36 * 64];
    __shared__ float h2[36 * 128];
    for (int i = threadIdx.x; i < 36 * 64; i += blockDim.x) {
        int r = i / 64, c = i % 64;
        float s = b1[c];
        for (int k = 0; k < 768; ++k) s = fmaf(tf[r * 768 + k], w1[k * 64 + c], s);
        h1[i] = fmaxf(s, 0.f);
    }
    __syncthreads();
    for (int i = threadIdx.x; i < 36 * 128; i += blockDim.x) {
        int r = i / 128, c = i % 128;
        float s = b2[c];
        for (int k = 0; k < 64; ++k) s = fmaf(h1[r * 64 + k], w2[k * 128 + c], s);
        h2[i] = fmaxf(s, 0.f);
    }
    __syncthreads();
    for (int i = threadIdx.x; i < 36 * 32; i += blockDim.x) {
        int r = i / 32, c = i % 32;
        float s = b3[c];
        for (int k = 0; k < 128; ++k) s = fmaf(h2[r * 128 + k], w3[k * 32 + c], s);
        baseOut[i] = s;
    }
}

// ---------------------------------------------------------------------------
// Fused conv(32->3, 3^3, pad 1) + gather/einsum + bias.
// One wave per 16-voxel W-tile; implicit GEMM via v_wmma_f32_16x16x32_bf16.
// ---------------------------------------------------------------------------
__global__ void __launch_bounds__(NW * 32)
fused_kernel(const float* __restrict__ x,      // (32,80,80,80)
             const int*   __restrict__ y_seg,  // (1,1,160,160,160)
             const float* __restrict__ baseT,  // (3,36,32) from MLPs
             const v16bf* __restrict__ Bops,   // (27,32) lane-packed B operands
             const float* __restrict__ conv_b, // (3)
             float*       __restrict__ out) {  // (3,80,80,80)
    __shared__ __attribute__((aligned(64))) __bf16 xs[NW][9][18][CCH];
    __shared__ float outs[NW][3][16];
    __shared__ float bases[3 * N_CLS * CCH];
    const int lane = threadIdx.x & 31;
    const int wave = threadIdx.x >> 5;

    __builtin_prefetch(Bops, 0, 1);   // global_prefetch_b8: pull B table toward L2/L0

    // stage the per-class direction vectors once per block
    for (int i = threadIdx.x; i < 3 * N_CLS * CCH; i += blockDim.x) bases[i] = baseT[i];

    const int tile = blockIdx.x * NW + wave;      // < 32000 by construction
    const int d    = tile / (DIM * NTILE_W);
    const int rem  = tile % (DIM * NTILE_W);
    const int h    = rem / NTILE_W;
    const int w0   = (rem % NTILE_W) * 16;

    // ---- stage x slab -> LDS as bf16, layout [tap_row][w][channel] ----
    for (int r = 0; r < 9; ++r) {
        const int dr = d + r / 3 - 1;
        const int hr = h + r % 3 - 1;
        const bool rowok = ((unsigned)dr < (unsigned)DIM) & ((unsigned)hr < (unsigned)DIM);
        for (int e = lane; e < 18 * CCH; e += 32) {
            const int c = e / 18, w = e % 18;
            const int wg = w0 + w - 1;
            float v = 0.f;
            if (rowok && (unsigned)wg < (unsigned)DIM)
                v = x[(((size_t)c * DIM + dr) * DIM + hr) * DIM + wg];
            xs[wave][r][w][c] = (__bf16)v;
        }
    }
    __syncthreads();

    // ---- implicit-GEMM conv: 27 x v_wmma_f32_16x16x32_bf16 ----
    v8f acc = {};
    const int hi = lane >> 4;
    const int m16 = lane & 15;
    for (int r = 0; r < 9; ++r) {
#pragma unroll
        for (int kw = 0; kw < 3; ++kw) {
            // A fragment: lane m = lane%16 -> voxel row, K chunks {8*hi..} and {8*hi+16..}
            const __bf16* ap = &xs[wave][r][m16 + kw][hi * 8];
            v8bf a0 = *(const v8bf*)(ap);
            v8bf a1 = *(const v8bf*)(ap + 16);
            v16bf a = __builtin_shufflevector(a0, a1, 0, 1, 2, 3, 4, 5, 6, 7,
                                              8, 9, 10, 11, 12, 13, 14, 15);
            v16bf b = Bops[(r * 3 + kw) * 32 + lane];
            acc = __builtin_amdgcn_wmma_f32_16x16x32_bf16(
                false, a, false, b, (short)0, acc, false, false);
        }
    }

    // ---- spill conv result (voxel m, oc n) to LDS ----
    const int oc = m16, grp = hi;
    if (oc < 3) {
#pragma unroll
        for (int rr = 0; rr < 8; ++rr)
            outs[wave][oc][rr + 8 * grp] = acc[rr];
    }
    __syncthreads();

    // ---- einsum (gathered direction vectors) + bias + store ----
    if (lane < 16) {
        const int m  = lane;
        const int wg = w0 + m;
        const size_t vox = ((size_t)d * DIM + h) * DIM + wg;
        const int seg = y_seg[((size_t)(2 * d) * 160 + 2 * h) * 160 + 2 * wg];
        float dot[3];
#pragma unroll
        for (int dir = 0; dir < 3; ++dir) dot[dir] = 0.f;
        for (int c = 0; c < CCH; ++c) {
            const float xv = x[(size_t)c * (DIM * DIM * DIM) + vox];
#pragma unroll
            for (int dir = 0; dir < 3; ++dir)
                dot[dir] = fmaf(bases[(dir * N_CLS + seg) * CCH + c], xv, dot[dir]);
        }
#pragma unroll
        for (int dir = 0; dir < 3; ++dir)
            out[(size_t)dir * (DIM * DIM * DIM) + vox] =
                outs[wave][dir][m] + conv_b[dir] + dot[dir];
    }
}

extern "C" void kernel_launch(void* const* d_in, const int* in_sizes, int n_in,
                              void* d_out, int out_size, void* d_ws, size_t ws_size,
                              hipStream_t stream) {
    (void)in_sizes; (void)n_in; (void)out_size; (void)ws_size;
    const float* x       = (const float*)d_in[0];
    const int*   y_seg   = (const int*)d_in[1];
    const float* tf      = (const float*)d_in[2];
    const float* conv_w  = (const float*)d_in[21];
    const float* conv_b  = (const float*)d_in[22];

    float*  ws_base = (float*)d_ws;                        // 3*36*32 f32 = 13824 B
    __bf16* ws_B    = (__bf16*)((char*)d_ws + 16384);      // 27*32*16 bf16 = 27648 B

    pack_b_kernel<<<(27 * 32 * 16 + 255) / 256, 256, 0, stream>>>(conv_w, ws_B);

    for (int dir = 0; dir < 3; ++dir) {
        const float* w1 = (const float*)d_in[3 + dir * 6 + 0];
        const float* b1 = (const float*)d_in[3 + dir * 6 + 1];
        const float* w2 = (const float*)d_in[3 + dir * 6 + 2];
        const float* b2 = (const float*)d_in[3 + dir * 6 + 3];
        const float* w3 = (const float*)d_in[3 + dir * 6 + 4];
        const float* b3 = (const float*)d_in[3 + dir * 6 + 5];
        mlp_kernel<<<1, 256, 0, stream>>>(tf, w1, b1, w2, b2, w3, b3,
                                          ws_base + dir * 36 * 32);
    }

    fused_kernel<<<NTILES / NW, NW * 32, 0, stream>>>(
        x, y_seg, ws_base, (const v16bf*)ws_B, conv_b, (float*)d_out);
}
